// Refine_36197984371094
// MI455X (gfx1250) — matrix operand order repaired and checked by reading, not compiled, not run on hardware
//
#include <hip/hip_runtime.h>
#include <hip/hip_bf16.h>

typedef __attribute__((ext_vector_type(16))) __bf16 bf16x16;
typedef __attribute__((ext_vector_type(8)))  __bf16 bf16x8;
typedef __attribute__((ext_vector_type(8)))  float  v8f;
typedef int v4i_g __attribute__((vector_size(4 * sizeof(int))));

#define AS1 __attribute__((address_space(1)))
#define AS3 __attribute__((address_space(3)))

#if __has_builtin(__builtin_amdgcn_global_load_async_to_lds_b128) && \
    __has_builtin(__builtin_amdgcn_s_wait_asynccnt)
#define USE_ASYNC 1
#define ASYNC_WAIT(n) __builtin_amdgcn_s_wait_asynccnt(n)
#else
#define USE_ASYNC 0
#define ASYNC_WAIT(n)
#endif

#define BB    8
#define CIN   64
#define HH    128
#define WW    128
#define C1    256          // conv1 out channels
#define C2    64           // conv2 out channels
#define K1    576          // 9*64 im2col depth
#define NPOLY 8192
#define NPTS  129          // 1 ct point + 128 init points
#define KFP   8256         // 64*129
#define NX    512
#define NOUT  256

static __device__ __forceinline__ bf16x8 bzero8() {
  bf16x8 z;
#pragma unroll
  for (int i = 0; i < 8; ++i) z[i] = (__bf16)0.0f;
  return z;
}

static __device__ __forceinline__ bf16x16 cat8(bf16x8 lo, bf16x8 hi) {
  return __builtin_shufflevector(lo, hi, 0, 1, 2, 3, 4, 5, 6, 7,
                                         8, 9, 10, 11, 12, 13, 14, 15);
}

static __device__ __forceinline__ v8f wmma_bf16(bf16x16 a, bf16x16 b, v8f c) {
  return __builtin_amdgcn_wmma_f32_16x16x32_bf16(false, a, false, b,
                                                 (short)0, c, false, false);
}

// Stage one 4KB B slab (2048 bf16) into LDS: each of 256 threads moves 16B.
static __device__ __forceinline__ void stage_slab(const __bf16* gsrc,
                                                  __bf16* lds_dst, int tid) {
#if USE_ASYNC
  __bf16* g = const_cast<__bf16*>(gsrc) + tid * 8;
  __builtin_amdgcn_global_load_async_to_lds_b128(
      (AS1 v4i_g*)g, (AS3 v4i_g*)(lds_dst + tid * 8), 0, 0);
#else
  *(uint4*)(lds_dst + tid * 8) = *(const uint4*)(gsrc + tid * 8);
#endif
}

// ---------------------------------------------------------------------------
// feature (B,C,H,W) f32  ->  featn (B,H,W,C) bf16  (A-fragment friendly NHWC)
// ---------------------------------------------------------------------------
__global__ void cast_feat_nhwc(const float* __restrict__ feat,
                               __bf16* __restrict__ out) {
  int tid = blockIdx.x * 256 + threadIdx.x;
  if (tid >= BB * CIN * HH * WW) return;
  int c = tid & 63;
  int x = (tid >> 6) & 127;
  int y = (tid >> 13) & 127;
  int b = tid >> 20;
  out[tid] = (__bf16)feat[(((size_t)(b * CIN + c)) * HH + y) * WW + x];
}

// ---------------------------------------------------------------------------
// Pack a (N,K) f32 weight (B[k,n] = src[n*K+k]) into WMMA B fragments:
//   dst[((k/32)*(N/16) + n/16)*512 + lane*16 + j]
//   lane = ((k>>4)&1)*16 + (n&15), j = k&15
// ---------------------------------------------------------------------------
__global__ void pack_b_frag(const float* __restrict__ src,
                            __bf16* __restrict__ dst, int K, int N) {
  int idx = blockIdx.x * 256 + threadIdx.x;
  if (idx >= K * N) return;
  int k = idx / N;
  int n = idx - k * N;
  float v = src[(size_t)n * K + k];
  int kt = k >> 5, nt = n >> 4;
  int lane = ((k >> 4) & 1) * 16 + (n & 15);
  int j = k & 15;
  dst[((size_t)kt * (N >> 4) + nt) * 512 + lane * 16 + j] = (__bf16)v;
}

// w1 (256,64,3,3) -> packed B frags with k = (ky*3+kx)*64 + c, n = o, N=256
__global__ void pack_w1(const float* __restrict__ w1, __bf16* __restrict__ dst) {
  int idx = blockIdx.x * 256 + threadIdx.x;
  if (idx >= K1 * C1) return;
  int k = idx >> 8;          // / 256
  int n = idx & 255;
  int r = k >> 6;            // (ky*3+kx)
  int c = k & 63;
  int ky = r / 3, kx = r % 3;
  float v = w1[(((size_t)n * CIN + c) * 3 + ky) * 3 + kx];
  int kt = k >> 5, nt = n >> 4;
  int lane = ((k >> 4) & 1) * 16 + (n & 15);
  int j = k & 15;
  dst[((size_t)kt * (C1 >> 4) + nt) * 512 + lane * 16 + j] = (__bf16)v;
}

// ---------------------------------------------------------------------------
// conv1: implicit-im2col GEMM. M = B*H*W (131072), N = 256, K = 576.
// 8 waves/block, 16(M)x64(N) per wave; B slabs double-buffered in LDS via
// async global->LDS copies.
// ---------------------------------------------------------------------------
__global__ void conv1_wmma(const __bf16* __restrict__ featn,
                           const __bf16* __restrict__ w1p,
                           const float* __restrict__ b1,
                           __bf16* __restrict__ f1) {
  __shared__ __bf16 Bs[2][2048];
  const int tid = threadIdx.x;
  const int lane = tid & 31;
  const int wave = tid >> 5;
  const int half = lane >> 4;
  const int m0 = blockIdx.x * 128 + wave * 16;
  const int n0 = blockIdx.y * 64;
  const int nt0 = n0 >> 4;
  const int m = m0 + (lane & 15);
  const int b = m >> 14;
  const int y = (m >> 7) & 127;
  const int x = m & 127;
  const int KT = K1 >> 5;  // 18

  v8f zero = {};
  v8f acc[4];
#pragma unroll
  for (int t = 0; t < 4; ++t) acc[t] = zero;

  stage_slab(w1p + ((size_t)0 * (C1 >> 4) + nt0) * 512, &Bs[0][0], tid);

  for (int kt = 0; kt < KT; ++kt) {
    if (kt + 1 < KT) {
      stage_slab(w1p + ((size_t)(kt + 1) * (C1 >> 4) + nt0) * 512,
                 &Bs[(kt + 1) & 1][0], tid);
      ASYNC_WAIT(1);
    } else {
      ASYNC_WAIT(0);
    }
    __syncthreads();

    const int k0 = kt << 5;
    const int r = k0 >> 6;          // filter tap (ky*3+kx)
    const int c0 = k0 & 63;         // channel base (0 or 32)
    const int ky = r / 3, kx = r % 3;
    const int yy = y + ky - 1, xx = x + kx - 1;
    bf16x16 a;
    if (yy >= 0 && yy < HH && xx >= 0 && xx < WW) {
      const __bf16* p = featn + (((size_t)((b * HH + yy) * WW + xx)) << 6)
                        + c0 + half * 16;
      a = cat8(*(const bf16x8*)p, *(const bf16x8*)(p + 8));
    } else {
      bf16x8 z = bzero8();
      a = cat8(z, z);
    }

    const __bf16* bs = &Bs[kt & 1][lane * 16];
    bf16x16 bfrag[4];
#pragma unroll
    for (int t = 0; t < 4; ++t) bfrag[t] = *(const bf16x16*)(bs + t * 512);
#pragma unroll
    for (int t = 0; t < 4; ++t) acc[t] = wmma_bf16(a, bfrag[t], acc[t]);

    __syncthreads();
  }

#pragma unroll
  for (int t = 0; t < 4; ++t) {
    int col = n0 + t * 16 + (lane & 15);
    float bias = b1[col];
#pragma unroll
    for (int rr = 0; rr < 8; ++rr) {
      int row = m0 + half * 8 + rr;
      float v = acc[t][rr] + bias;
      f1[(size_t)row * C1 + col] = (__bf16)(v > 0.0f ? v : 0.0f);
    }
  }
}

// ---------------------------------------------------------------------------
// Generic WMMA GEMM: C(bf16) = optrelu(A(MxK bf16 row-major) * Bp + bias)
// Block tile 128(M) x 64(N); LDS double-buffered B slabs (async).
// ---------------------------------------------------------------------------
__global__ void gemm_wmma(const __bf16* __restrict__ A, int lda,
                          const __bf16* __restrict__ Bp,
                          const float* __restrict__ bias,
                          __bf16* __restrict__ C,
                          int N, int K, int relu) {
  __shared__ __bf16 Bs[2][2048];
  const int tid = threadIdx.x;
  const int lane = tid & 31;
  const int wave = tid >> 5;
  const int half = lane >> 4;
  const int m0 = blockIdx.x * 128 + wave * 16;
  const int n0 = blockIdx.y * 64;
  const int nt0 = n0 >> 4;
  const int ntiles = N >> 4;
  const int KT = K >> 5;
  const int m = m0 + (lane & 15);
  const __bf16* Arow = A + (size_t)m * lda + half * 16;

  v8f zero = {};
  v8f acc[4];
#pragma unroll
  for (int t = 0; t < 4; ++t) acc[t] = zero;

  stage_slab(Bp + ((size_t)0 * ntiles + nt0) * 512, &Bs[0][0], tid);

  for (int kt = 0; kt < KT; ++kt) {
    if (kt + 1 < KT) {
      stage_slab(Bp + ((size_t)(kt + 1) * ntiles + nt0) * 512,
                 &Bs[(kt + 1) & 1][0], tid);
      ASYNC_WAIT(1);
    } else {
      ASYNC_WAIT(0);
    }
    __syncthreads();

    const int k0 = kt << 5;
    bf16x16 a = cat8(*(const bf16x8*)(Arow + k0),
                     *(const bf16x8*)(Arow + k0 + 8));

    const __bf16* bs = &Bs[kt & 1][lane * 16];
    bf16x16 bfrag[4];
#pragma unroll
    for (int t = 0; t < 4; ++t) bfrag[t] = *(const bf16x16*)(bs + t * 512);
#pragma unroll
    for (int t = 0; t < 4; ++t) acc[t] = wmma_bf16(a, bfrag[t], acc[t]);

    __syncthreads();
  }

#pragma unroll
  for (int t = 0; t < 4; ++t) {
    int col = n0 + t * 16 + (lane & 15);
    float bv = bias ? bias[col] : 0.0f;
#pragma unroll
    for (int rr = 0; rr < 8; ++rr) {
      int row = m0 + half * 8 + rr;
      float v = acc[t][rr] + bv;
      if (relu) v = v > 0.0f ? v : 0.0f;
      C[(size_t)row * N + col] = (__bf16)v;
    }
  }
}

// ---------------------------------------------------------------------------
// Bilinear sampling of f2 (NHWC bf16, C=64) at 129 points per poly.
// fp[i, c*129 + p]  (matches reference transpose(0,2,1).reshape)
// ---------------------------------------------------------------------------
__global__ void sample_points(const __bf16* __restrict__ f2,
                              const float* __restrict__ ct,
                              const float* __restrict__ init,
                              const int* __restrict__ imgidx,
                              __bf16* __restrict__ fp) {
  int tid = blockIdx.x * 256 + threadIdx.x;
  if (tid >= NPOLY * NPTS) return;
  int i = tid / NPTS;
  int p = tid - i * NPTS;

  float X, Y;
  if (p == 0) {
    X = ct[2 * i];
    Y = ct[2 * i + 1];
  } else {
    const float* q = init + ((size_t)i * 128 + (p - 1)) * 2;
    X = q[0];
    Y = q[1];
  }
  float px = X - 0.5f, py = Y - 0.5f;
  float x0f = floorf(px), y0f = floorf(py);
  float wx1 = px - x0f, wy1 = py - y0f;
  float wx0 = 1.0f - wx1, wy0 = 1.0f - wy1;
  int b = imgidx[i];

  float xs[2] = {x0f, x0f + 1.0f};
  float ys[2] = {y0f, y0f + 1.0f};
  float wxs[2] = {wx0, wx1};
  float wys[2] = {wy0, wy1};

  const __bf16* base[4];
  float wgt[4];
#pragma unroll
  for (int cy = 0; cy < 2; ++cy) {
#pragma unroll
    for (int cx = 0; cx < 2; ++cx) {
      float xc = xs[cx], yc = ys[cy];
      bool valid = (xc >= 0.0f) && (xc <= (float)(WW - 1)) &&
                   (yc >= 0.0f) && (yc <= (float)(HH - 1));
      int xi = (int)fminf(fmaxf(xc, 0.0f), (float)(WW - 1));
      int yi = (int)fminf(fmaxf(yc, 0.0f), (float)(HH - 1));
      int q = cy * 2 + cx;
      base[q] = f2 + (((size_t)((b * HH + yi) * WW + xi)) << 6);
      wgt[q] = wxs[cx] * wys[cy] * (valid ? 1.0f : 0.0f);
    }
  }

  __bf16* outp = fp + (size_t)i * KFP + p;
#pragma unroll 8
  for (int c = 0; c < 64; ++c) {
    float v = wgt[0] * (float)base[0][c] + wgt[1] * (float)base[1][c] +
              wgt[2] * (float)base[2][c] + wgt[3] * (float)base[3][c];
    outp[(size_t)c * NPTS] = (__bf16)v;
  }
}

// ---------------------------------------------------------------------------
// Final GEMM with fused epilogue:
// out[i,n] = (x[i,:] . wf[n,:] + bf[n]) * 4.0 + init_polys[i,n]
// M = 8192, K = 512, N = 256, f32 output straight to d_out.
// ---------------------------------------------------------------------------
__global__ void gemm_final(const __bf16* __restrict__ X,
                           const __bf16* __restrict__ Bp,
                           const float* __restrict__ bias,
                           const float* __restrict__ init,
                           float* __restrict__ out) {
  __shared__ __bf16 Bs[2][2048];
  const int tid = threadIdx.x;
  const int lane = tid & 31;
  const int wave = tid >> 5;
  const int half = lane >> 4;
  const int m0 = blockIdx.x * 128 + wave * 16;
  const int n0 = blockIdx.y * 64;
  const int nt0 = n0 >> 4;
  const int KT = NX >> 5;  // 16
  const int m = m0 + (lane & 15);
  const __bf16* Arow = X + (size_t)m * NX + half * 16;

  v8f zero = {};
  v8f acc[4];
#pragma unroll
  for (int t = 0; t < 4; ++t) acc[t] = zero;

  stage_slab(Bp + ((size_t)0 * (NOUT >> 4) + nt0) * 512, &Bs[0][0], tid);

  for (int kt = 0; kt < KT; ++kt) {
    if (kt + 1 < KT) {
      stage_slab(Bp + ((size_t)(kt + 1) * (NOUT >> 4) + nt0) * 512,
                 &Bs[(kt + 1) & 1][0], tid);
      ASYNC_WAIT(1);
    } else {
      ASYNC_WAIT(0);
    }
    __syncthreads();

    const int k0 = kt << 5;
    bf16x16 a = cat8(*(const bf16x8*)(Arow + k0),
                     *(const bf16x8*)(Arow + k0 + 8));

    const __bf16* bs = &Bs[kt & 1][lane * 16];
    bf16x16 bfrag[4];
#pragma unroll
    for (int t = 0; t < 4; ++t) bfrag[t] = *(const bf16x16*)(bs + t * 512);
#pragma unroll
    for (int t = 0; t < 4; ++t) acc[t] = wmma_bf16(a, bfrag[t], acc[t]);

    __syncthreads();
  }

#pragma unroll
  for (int t = 0; t < 4; ++t) {
    int col = n0 + t * 16 + (lane & 15);
    float bv = bias[col];
#pragma unroll
    for (int rr = 0; rr < 8; ++rr) {
      int row = m0 + half * 8 + rr;
      size_t o = (size_t)row * NOUT + col;
      out[o] = (acc[t][rr] + bv) * 4.0f + init[o];
    }
  }
}

// ---------------------------------------------------------------------------
extern "C" void kernel_launch(void* const* d_in, const int* in_sizes, int n_in,
                              void* d_out, int out_size, void* d_ws,
                              size_t ws_size, hipStream_t stream) {
  const float* feature = (const float*)d_in[0];
  const float* ct      = (const float*)d_in[1];
  const float* init    = (const float*)d_in[2];
  const int*   imgidx  = (const int*)d_in[3];
  const float* w1      = (const float*)d_in[4];
  const float* b1      = (const float*)d_in[5];
  const float* w2      = (const float*)d_in[6];
  const float* b2      = (const float*)d_in[7];
  const float* wp      = (const float*)d_in[8];
  const float* wf      = (const float*)d_in[9];
  const float* bf      = (const float*)d_in[10];
  float* out = (float*)d_out;

  char* ws = (char*)d_ws;
  size_t off = 0;
  auto alloc = [&](size_t bytes) -> void* {
    void* p = ws + off;
    off = (off + bytes + 255) & ~(size_t)255;
    return p;
  };

  __bf16* featn = (__bf16*)alloc((size_t)BB * HH * WW * CIN * 2);
  __bf16* w1p   = (__bf16*)alloc((size_t)K1 * C1 * 2);
  __bf16* w2p   = (__bf16*)alloc((size_t)C1 * C2 * 2);
  __bf16* wpP   = (__bf16*)alloc((size_t)KFP * NX * 2);
  __bf16* wfP   = (__bf16*)alloc((size_t)NX * NOUT * 2);
  __bf16* f1    = (__bf16*)alloc((size_t)BB * HH * WW * C1 * 2);
  __bf16* f2    = (__bf16*)alloc((size_t)BB * HH * WW * C2 * 2);
  __bf16* fp    = (__bf16*)alloc((size_t)NPOLY * KFP * 2);
  __bf16* xbuf  = (__bf16*)alloc((size_t)NPOLY * NX * 2);

  const int M = BB * HH * WW;  // 131072

  int nfeat = BB * CIN * HH * WW;
  cast_feat_nhwc<<<(nfeat + 255) / 256, 256, 0, stream>>>(feature, featn);
  pack_w1<<<(K1 * C1 + 255) / 256, 256, 0, stream>>>(w1, w1p);
  pack_b_frag<<<(C1 * C2 + 255) / 256, 256, 0, stream>>>(w2, w2p, C1, C2);
  pack_b_frag<<<(KFP * NX + 255) / 256, 256, 0, stream>>>(wp, wpP, KFP, NX);
  pack_b_frag<<<(NX * NOUT + 255) / 256, 256, 0, stream>>>(wf, wfP, NX, NOUT);

  // conv1 (3x3, 64->256) + ReLU
  conv1_wmma<<<dim3(M / 128, C1 / 64), 256, 0, stream>>>(featn, w1p, b1, f1);
  // conv2 (1x1, 256->64) + bias
  gemm_wmma<<<dim3(M / 128, C2 / 64), 256, 0, stream>>>(f1, C1, w2p, b2, f2,
                                                        C2, C1, 0);
  // bilinear sampling -> fp (8192 x 8256)
  sample_points<<<(NPOLY * NPTS + 255) / 256, 256, 0, stream>>>(f2, ct, init,
                                                                imgidx, fp);
  // x = fp @ wp.T   (8192 x 512, K = 8256)
  gemm_wmma<<<dim3(NPOLY / 128, NX / 64), 256, 0, stream>>>(
      fp, KFP, wpP, nullptr, xbuf, NX, KFP, 0);
  // out = (x @ wf.T + bf) * 4 + init_polys
  gemm_final<<<dim3(NPOLY / 128, NOUT / 64), 256, 0, stream>>>(xbuf, wfP, bf,
                                                               init, out);
}
